// SelfAttention_34041910788246
// MI455X (gfx1250) — matrix operand order repaired and checked by reading, not compiled, not run on hardware
//
#include <hip/hip_runtime.h>
#include <math.h>

#define BATCH  4
#define HEADS  16
#define SEQ    1024
#define HDIM   64
#define QTILE  128
#define KTILE  128
#define NWAVES 8
#define NKT    (SEQ / KTILE)
#define NQT    (SEQ / QTILE)

typedef _Float16 half_t;
typedef __attribute__((ext_vector_type(16))) _Float16     v16h;
typedef __attribute__((ext_vector_type(8)))  float        v8f;
typedef __attribute__((ext_vector_type(4)))  unsigned int u32x4;
typedef __attribute__((ext_vector_type(2)))  unsigned int u32x2;
typedef __attribute__((ext_vector_type(4)))  int          i32x4;

#if defined(__has_builtin)
#if __has_builtin(__builtin_amdgcn_global_load_async_to_lds_b128) && \
    __has_builtin(__builtin_amdgcn_s_wait_asynccnt)
#define USE_ASYNC 1
#endif
#endif
#ifndef USE_ASYNC
#define USE_ASYNC 0
#endif

// Exact parameter types for the async builtin (from clang's diagnostic):
// param 1: int __vector(4) in address space 1; param 2: int __vector(4) in LDS.
typedef int v4i __attribute__((vector_size(16)));
typedef __attribute__((address_space(1))) v4i* gv4i_p;
typedef __attribute__((address_space(3))) v4i* lv4i_p;

union HV {
    v16h  h;
    u32x4 q[2];
};

static __device__ __forceinline__ unsigned pack2(float a, float b) {
    half_t ha = (half_t)a, hb = (half_t)b;
    unsigned short ua = __builtin_bit_cast(unsigned short, ha);
    unsigned short ub = __builtin_bit_cast(unsigned short, hb);
    return (unsigned)ua | ((unsigned)ub << 16);
}

// ---------------------------------------------------------------------------
// Shared flash-attention tile core.
// sK: [key][dim] f16 row-major (KTILE x HDIM). sV: [dim][key] f16 (HDIM x KTILE).
// Scores computed as S^T = K x Q^T so softmax rows are lane-local.
// ---------------------------------------------------------------------------
static __device__ __forceinline__ void fa_core_tile(
    const half_t* sK, const half_t* sV, const v16h bq[2],
    const int* maskTile, float& m_run, float& l_run, v8f O[4], int lane)
{
    const bool hi  = lane >= 16;
    const int  l16 = lane & 15;
    const v8f  vz  = {0.f, 0.f, 0.f, 0.f, 0.f, 0.f, 0.f, 0.f};

    // ---- S^T = K_tile x Q^T
    v8f sc[8];
    #pragma unroll
    for (int m = 0; m < 8; ++m) {
        v8f acc = vz;
        #pragma unroll
        for (int cc = 0; cc < 2; ++cc) {
            HV a;
            const int off = (m * 16 + l16) * HDIM + cc * 32 + (hi ? 8 : 0);
            a.q[0] = *(const u32x4*)(&sK[off]);
            a.q[1] = *(const u32x4*)(&sK[off + 16]);
            acc = __builtin_amdgcn_wmma_f32_16x16x32_f16(
                false, a.h, false, bq[cc], (short)0, acc, false, false);
        }
        sc[m] = acc;
    }

    // ---- mask + row max (lane-local, then pair with lane^16)
    float tmax = -__builtin_inff();
    #pragma unroll
    for (int m = 0; m < 8; ++m) {
        const int* mp = maskTile + m * 16 + (hi ? 8 : 0);
        i32x4 mv0 = *(const i32x4*)(mp);
        i32x4 mv1 = *(const i32x4*)(mp + 4);
        #pragma unroll
        for (int r = 0; r < 8; ++r) {
            const int mv = (r < 4) ? mv0[r] : mv1[r - 4];
            float s = sc[m][r];
            s = (mv == 0) ? -1e9f : s;
            sc[m][r] = s;
            tmax = fmaxf(tmax, s);
        }
    }
    tmax = fmaxf(tmax, __shfl_xor(tmax, 16, 32));
    const float m_new = fmaxf(m_run, tmax);
    const float alpha = __expf(m_run - m_new);

    float lsum = 0.f;
    #pragma unroll
    for (int m = 0; m < 8; ++m) {
        #pragma unroll
        for (int r = 0; r < 8; ++r) {
            const float p = __expf(sc[m][r] - m_new);
            sc[m][r] = p;
            lsum += p;
        }
    }
    lsum += __shfl_xor(lsum, 16, 32);
    l_run = l_run * alpha + lsum;
    m_run = m_new;

    // ---- rescale O: vgpr r belongs to query (r + 8*hi); alpha lives in that lane
    #pragma unroll
    for (int r = 0; r < 8; ++r) {
        const float ar = __shfl(alpha, r + (hi ? 8 : 0), 32);
        #pragma unroll
        for (int nt = 0; nt < 4; ++nt) O[nt][r] *= ar;
    }

    // ---- O += P x V.  P (C-layout) repacks lane-locally into A-layout.
    #pragma unroll
    for (int cc = 0; cc < 4; ++cc) {
        v16h pA;
        #pragma unroll
        for (int j = 0; j < 8; ++j) {
            pA[2 * j]     = (half_t)sc[2 * cc + (j >> 2)][2 * (j & 3)];
            pA[2 * j + 1] = (half_t)sc[2 * cc + (j >> 2)][2 * (j & 3) + 1];
        }
        #pragma unroll
        for (int nt = 0; nt < 4; ++nt) {
            HV bv;
            const int off = (nt * 16 + l16) * KTILE + cc * 32 + (hi ? 8 : 0);
            bv.q[0] = *(const u32x4*)(&sV[off]);
            bv.q[1] = *(const u32x4*)(&sV[off + 16]);
            O[nt] = __builtin_amdgcn_wmma_f32_16x16x32_f16(
                false, pA, false, bv.h, (short)0, O[nt], false, false);
        }
    }
}

// Load Q rows into B-matrix layout, f16, pre-scaled by 1/sqrt(D).
static __device__ __forceinline__ void load_q(const float* Qrow, v16h bq[2], int lane)
{
    const bool  hi    = lane >= 16;
    const float scale = 0.125f;
    #pragma unroll
    for (int cc = 0; cc < 2; ++cc) {
        #pragma unroll
        for (int j = 0; j < 8; ++j) {
            const int row0 = 2 * j + (hi ? 8 : 0) + (j >= 4 ? 8 : 0);
            float2 f = *(const float2*)(Qrow + cc * 32 + row0);
            bq[cc][2 * j]     = (half_t)(f.x * scale);
            bq[cc][2 * j + 1] = (half_t)(f.y * scale);
        }
    }
}

// Epilogue: partial[d] = sum over this block's 128 queries of O[q][d]/l[q].
static __device__ __forceinline__ void fa_epilogue(
    const v8f O[4], float l_run, float* red, float* partialOut,
    int tid, int lane, int wave)
{
    const bool hi  = lane >= 16;
    const int  l16 = lane & 15;
    float part[4];
    #pragma unroll
    for (int nt = 0; nt < 4; ++nt) part[nt] = 0.f;
    #pragma unroll
    for (int r = 0; r < 8; ++r) {
        const float lr  = __shfl(l_run, r + (hi ? 8 : 0), 32);
        const float inv = 1.0f / lr;
        #pragma unroll
        for (int nt = 0; nt < 4; ++nt) part[nt] += O[nt][r] * inv;
    }
    #pragma unroll
    for (int nt = 0; nt < 4; ++nt) part[nt] += __shfl_xor(part[nt], 16, 32);

    __syncthreads();
    if (!hi) {
        #pragma unroll
        for (int nt = 0; nt < 4; ++nt) red[wave * HDIM + nt * 16 + l16] = part[nt];
    }
    __syncthreads();
    if (tid < HDIM) {
        float s = 0.f;
        #pragma unroll
        for (int w = 0; w < NWAVES; ++w) s += red[w * HDIM + tid];
        partialOut[tid] = s;
    }
}

// ---------------------------------------------------------------------------
// One-shot converter: K -> f16 row-major, V -> f16 transposed [bh][dim][key].
// ---------------------------------------------------------------------------
__global__ __launch_bounds__(256)
void convert_kernel(const float* __restrict__ K, const float* __restrict__ V,
                    half_t* __restrict__ Kh, half_t* __restrict__ Vt)
{
    const int idx = blockIdx.x * 256 + threadIdx.x;   // float4 index
    const int bh  = idx / (SEQ * HDIM / 4);
    const int f4  = idx % (SEQ * HDIM / 4);
    const int key = f4 >> 4;
    const int db  = (f4 & 15) * 4;
    const size_t base = (size_t)bh * SEQ * HDIM;

    float4 kf = *(const float4*)(K + base + key * HDIM + db);
    u32x2 kp = {pack2(kf.x, kf.y), pack2(kf.z, kf.w)};
    *(u32x2*)(Kh + base + key * HDIM + db) = kp;

    float4 vf = *(const float4*)(V + base + key * HDIM + db);
    const size_t vb = (size_t)bh * HDIM * SEQ;
    Vt[vb + (size_t)(db + 0) * SEQ + key] = (half_t)vf.x;
    Vt[vb + (size_t)(db + 1) * SEQ + key] = (half_t)vf.y;
    Vt[vb + (size_t)(db + 2) * SEQ + key] = (half_t)vf.z;
    Vt[vb + (size_t)(db + 3) * SEQ + key] = (half_t)vf.w;
}

// Stage one K/V tile (already f16, async byte copy if available).
// Per wave: 8 async b128 ops (4 K + 4 V) -> s_wait_asynccnt 8 overlaps 1 tile.
static __device__ __forceinline__ void stage_tile(
    half_t* KhTile, half_t* VtBase, int key0, half_t* sKb, half_t* sVb, int tid)
{
#if USE_ASYNC
    #pragma unroll
    for (int r = 0; r < 4; ++r) {
        const int off = (tid + r * 256) * 16;     // bytes within the 16KB K tile
        __builtin_amdgcn_global_load_async_to_lds_b128(
            (gv4i_p)(void*)((char*)KhTile + off),
            (lv4i_p)(void*)((char*)sKb + off), 0, 0);
    }
    #pragma unroll
    for (int r = 0; r < 4; ++r) {
        const int c  = tid + r * 256;             // 16-byte chunk id
        const int d  = c >> 4;                    // dim row
        const int xo = (c & 15) * 16;             // bytes within row
        __builtin_amdgcn_global_load_async_to_lds_b128(
            (gv4i_p)(void*)((char*)(VtBase + (size_t)d * SEQ + key0) + xo),
            (lv4i_p)(void*)((char*)sVb + d * 256 + xo), 0, 0);
    }
#else
    #pragma unroll
    for (int r = 0; r < 4; ++r) {
        const int off = (tid + r * 256) * 16;
        *(u32x4*)((char*)sKb + off) = *(const u32x4*)((const char*)KhTile + off);
    }
    #pragma unroll
    for (int r = 0; r < 4; ++r) {
        const int c  = tid + r * 256;
        const int d  = c >> 4;
        const int xo = (c & 15) * 16;
        *(u32x4*)((char*)sVb + d * 256 + xo) =
            *(const u32x4*)((const char*)(VtBase + (size_t)d * SEQ + key0) + xo);
    }
#endif
}

static __device__ __forceinline__ void stage_wait_prev()
{
#if USE_ASYNC
    __builtin_amdgcn_s_wait_asynccnt(8);
#endif
}
static __device__ __forceinline__ void stage_wait_all()
{
#if USE_ASYNC
    __builtin_amdgcn_s_wait_asynccnt(0);
#endif
}

// ---------------------------------------------------------------------------
// Fancy path: f16 K/V from workspace, double-buffered async LDS staging.
// ---------------------------------------------------------------------------
__global__ __launch_bounds__(256)
void fa_sum_async(const float* __restrict__ Qg, half_t* __restrict__ Kh,
                  half_t* __restrict__ Vt, const int* __restrict__ Mg,
                  float* __restrict__ partials)
{
    __shared__ half_t sK[2][KTILE * HDIM];   // 2 x 16KB
    __shared__ half_t sV[2][HDIM * KTILE];   // 2 x 16KB

    const int tid  = threadIdx.x;
    const int lane = tid & 31;
    const int wave = tid >> 5;

    const int bh = blockIdx.x;
    const int qt = blockIdx.y;
    const int b  = bh / HEADS;

    const size_t baseQKV = (size_t)bh * SEQ * HDIM;
    const int    q_abs   = qt * QTILE + wave * 16 + (lane & 15);

    v16h bq[2];
    load_q(Qg + baseQKV + (size_t)q_abs * HDIM, bq, lane);

    const int* maskRow = Mg + ((size_t)b * SEQ + q_abs) * SEQ;
    half_t* KhB = Kh + baseQKV;
    half_t* VtB = Vt + (size_t)bh * HDIM * SEQ;

    float m_run = -__builtin_inff();
    float l_run = 0.0f;
    const v8f vz = {0.f, 0.f, 0.f, 0.f, 0.f, 0.f, 0.f, 0.f};
    v8f O[4];
    #pragma unroll
    for (int nt = 0; nt < 4; ++nt) O[nt] = vz;

    // Prologue: start tile 0 into buffer 0.
    stage_tile(KhB, VtB, 0, sK[0], sV[0], tid);

    #pragma unroll 1
    for (int kt = 0; kt < NKT; ++kt) {
        __syncthreads();   // everyone done with the buffer we are about to refill
        if (kt + 1 < NKT) {
            stage_tile(KhB + (size_t)(kt + 1) * KTILE * HDIM, VtB,
                       (kt + 1) * KTILE, sK[(kt + 1) & 1], sV[(kt + 1) & 1], tid);
            stage_wait_prev();   // tile kt complete (in-order), tile kt+1 in flight
        } else {
            stage_wait_all();
        }
        __syncthreads();   // tile kt visible to all waves

        fa_core_tile(sK[kt & 1], sV[kt & 1], bq, maskRow + kt * KTILE,
                     m_run, l_run, O, lane);
    }

    // Reuse K buffer 0 as the cross-wave reduction scratch (last tile used buf 1).
    fa_epilogue(O, l_run, (float*)sK[0],
                partials + ((size_t)bh * NQT + qt) * HDIM, tid, lane, wave);
}

// ---------------------------------------------------------------------------
// Fallback path: fp32 inputs, synchronous staging with in-register conversion.
// ---------------------------------------------------------------------------
__global__ __launch_bounds__(256)
void fa_sum_direct(const float* __restrict__ Qg, const float* __restrict__ Kg0,
                   const float* __restrict__ Vg0, const int* __restrict__ Mg,
                   float* __restrict__ partials)
{
    __shared__ half_t sK[KTILE * HDIM];
    __shared__ half_t sV[HDIM * KTILE];
    __shared__ float  red[NWAVES * HDIM];

    const int tid  = threadIdx.x;
    const int lane = tid & 31;
    const int wave = tid >> 5;

    const int bh = blockIdx.x;
    const int qt = blockIdx.y;
    const int b  = bh / HEADS;

    const size_t baseQKV = (size_t)bh * SEQ * HDIM;
    const int    q_abs   = qt * QTILE + wave * 16 + (lane & 15);

    v16h bq[2];
    load_q(Qg + baseQKV + (size_t)q_abs * HDIM, bq, lane);

    const int* maskRow = Mg + ((size_t)b * SEQ + q_abs) * SEQ;

    float m_run = -__builtin_inff();
    float l_run = 0.0f;
    const v8f vz = {0.f, 0.f, 0.f, 0.f, 0.f, 0.f, 0.f, 0.f};
    v8f O[4];
    #pragma unroll
    for (int nt = 0; nt < 4; ++nt) O[nt] = vz;

    #pragma unroll 1
    for (int kt = 0; kt < NKT; ++kt) {
        const float* Kg = Kg0 + baseQKV + (size_t)kt * KTILE * HDIM;
        const float* Vg = Vg0 + baseQKV + (size_t)kt * KTILE * HDIM;

        __syncthreads();
        #pragma unroll
        for (int t = 0; t < 8; ++t) {
            const int idx = tid + t * 256;
            const int kl  = idx >> 4;
            const int db  = (idx & 15) * 4;
            float4 kf = *(const float4*)(Kg + kl * HDIM + db);
            u32x2 kp = {pack2(kf.x, kf.y), pack2(kf.z, kf.w)};
            *(u32x2*)(&sK[kl * HDIM + db]) = kp;
            float4 vf = *(const float4*)(Vg + kl * HDIM + db);
            sV[(db + 0) * KTILE + kl] = (half_t)vf.x;
            sV[(db + 1) * KTILE + kl] = (half_t)vf.y;
            sV[(db + 2) * KTILE + kl] = (half_t)vf.z;
            sV[(db + 3) * KTILE + kl] = (half_t)vf.w;
            if (kt + 1 < NKT) {
                __builtin_prefetch(Kg + KTILE * HDIM + kl * HDIM + db, 0, 1);
                __builtin_prefetch(Vg + KTILE * HDIM + kl * HDIM + db, 0, 1);
            }
        }
        __syncthreads();

        fa_core_tile(sK, sV, bq, maskRow + kt * KTILE, m_run, l_run, O, lane);
    }

    fa_epilogue(O, l_run, red, partials + ((size_t)bh * NQT + qt) * HDIM,
                tid, lane, wave);
}

// Deterministic ordered reduction of the 8 query-tile partials per (b,h).
__global__ __launch_bounds__(256)
void fa_reduce_kernel(const float* __restrict__ partials, float* __restrict__ out)
{
    const int i  = blockIdx.x * blockDim.x + threadIdx.x;   // 0..4095
    const int bh = i >> 6;
    const int d  = i & 63;
    float s = 0.f;
    #pragma unroll
    for (int qt = 0; qt < NQT; ++qt)
        s += partials[((size_t)bh * NQT + qt) * HDIM + d];
    out[i] = s;
}

extern "C" void kernel_launch(void* const* d_in, const int* in_sizes, int n_in,
                              void* d_out, int out_size, void* d_ws, size_t ws_size,
                              hipStream_t stream) {
    const float* q    = (const float*)d_in[0];
    const float* k    = (const float*)d_in[1];
    const float* v    = (const float*)d_in[2];
    const int*   mask = (const int*)d_in[3];

    const size_t KH_BYTES   = (size_t)BATCH * HEADS * SEQ * HDIM * sizeof(half_t); // 8 MB
    const size_t VT_BYTES   = KH_BYTES;                                            // 8 MB
    const size_t PART_BYTES = (size_t)BATCH * HEADS * NQT * HDIM * sizeof(float);  // 128 KB

    dim3 grid(BATCH * HEADS, NQT);
    const int nOut = BATCH * HEADS * HDIM;   // 4096

    if (ws_size >= KH_BYTES + VT_BYTES + PART_BYTES) {
        half_t* Kh   = (half_t*)d_ws;
        half_t* Vt   = (half_t*)((char*)d_ws + KH_BYTES);
        float*  part = (float*)((char*)d_ws + KH_BYTES + VT_BYTES);

        const int nConv = BATCH * HEADS * SEQ * HDIM / 4;   // 1M float4s
        convert_kernel<<<nConv / 256, 256, 0, stream>>>(k, v, Kh, Vt);
        fa_sum_async<<<grid, 256, 0, stream>>>(q, Kh, Vt, mask, part);
        fa_reduce_kernel<<<(nOut + 255) / 256, 256, 0, stream>>>(part, (float*)d_out);
    } else {
        float* part = (float*)d_ws;   // needs 128 KB
        fa_sum_direct<<<grid, 256, 0, stream>>>(q, k, v, mask, part);
        fa_reduce_kernel<<<(nOut + 255) / 256, 256, 0, stream>>>(part, (float*)d_out);
    }
}
